// GCNNet_62423054680283
// MI455X (gfx1250) — matrix operand order repaired and checked by reading, not compiled, not run on hardware
//
#include <hip/hip_runtime.h>

// CDNA5 / gfx1250: wave32, WMMA 16x16x32 bf16.
typedef __attribute__((ext_vector_type(16))) __bf16 v16bf;
typedef __attribute__((ext_vector_type(8)))  __bf16 v8bf;
typedef __attribute__((ext_vector_type(8)))  float  v8f;

__device__ __forceinline__ void atomic_add_f32(float* p, float v) {
    __hip_atomic_fetch_add(p, v, __ATOMIC_RELAXED, __HIP_MEMORY_SCOPE_AGENT);
}

// ---------------------------------------------------------------------------
// Degree accumulation: deg[dst[e]] += 1 per edge.
// ---------------------------------------------------------------------------
__global__ void degree_kernel(const long long* __restrict__ dst, float* __restrict__ deg, int E) {
    int t = blockIdx.x * blockDim.x + threadIdx.x;
    if (t < E) atomic_add_f32(&deg[(int)dst[t]], 1.0f);
}

// dinv[i] = rsqrt(deg[i] + 1)   (in place: deg buffer becomes dinv)
__global__ void dinv_kernel(float* __restrict__ deg, int n) {
    int t = blockIdx.x * blockDim.x + threadIdx.x;
    if (t < n) deg[t] = rsqrtf(deg[t] + 1.0f);
}

// ---------------------------------------------------------------------------
// Pack W (K x NCOLS f32, row-major) into bf16 WMMA B-fragment order:
//   element linear index t = ((kbi*NT + nt)*2 + h)*256 + lane*8 + j
//   holds W[k][col] with k = kbi*32 + h*16 + (lane/16)*8 + j,
//                       col = nt*16 + (lane%16).
// Each (fragment, half) is 512 B with lane-stride 16 B -> conflict-free
// ds_load_b128 when staged in LDS.
// ---------------------------------------------------------------------------
template <int NT>
__global__ void pack_w(const float* __restrict__ W, __bf16* __restrict__ P, int K) {
    constexpr int NCOLS = NT * 16;
    int t = blockIdx.x * blockDim.x + threadIdx.x;
    if (t >= K * NCOLS) return;
    int j    = t & 7;
    int lane = (t >> 3) & 31;
    int h    = (t >> 8) & 1;
    int rest = t >> 9;                 // fragment index = kbi*NT + nt
    int nt   = rest % NT;
    int kbi  = rest / NT;
    int k    = kbi * 32 + h * 16 + (lane >> 4) * 8 + j;
    int col  = nt * 16 + (lane & 15);
    P[t] = (__bf16)W[(size_t)k * NCOLS + col];
}

// ---------------------------------------------------------------------------
// WMMA GEMM:  C[M x NCOLS] = A[M x K] * W[K x NCOLS]  (f32 in/out, bf16
// multiply, f32 accumulate).  K, NCOLS compile-time -> fully unrolled.
// One wave per 16-row M-tile, accumulating all N-tiles (A loaded once per
// K-step).  Packed bf16 W is staged in LDS once per block; fragment reads are
// conflict-free ds_load_b128 pairs instead of 16 strided VMEM loads.
// Fragment layouts per CDNA5 ISA 7.12.2 (wave32):
//   A (16x32, 16-bit): lane l holds row M=l%16; elems 0..7  -> K=kb+8g+0..7,
//                      elems 8..15 -> K=kb+16+8g+0..7   (g = l/16)
//   B (32x16, 16-bit): lane l holds col N=l%16; elem i -> same K as A's elem i
//   C/D (16x16 f32):   lane l, vgpr r -> M = 8*(l/16)+r, N = l%16
// ---------------------------------------------------------------------------
template <int K, int NCOLS>
__global__ void gemm_wmma_bf16(const float* __restrict__ A, const __bf16* __restrict__ Wp,
                               float* __restrict__ C, int nMTiles) {
    constexpr int NT  = NCOLS / 16;
    constexpr int KT  = K / 32;
    constexpr int ELEMS = K * NCOLS;               // packed bf16 elements

    __shared__ alignas(16) __bf16 sW[ELEMS];

    // Cooperative fill: contiguous 16B chunks, coalesced.
    {
        const int4* gp = (const int4*)Wp;
        int4*       sp = (int4*)sW;
        for (int i = threadIdx.x; i < ELEMS / 8; i += blockDim.x) sp[i] = gp[i];
    }
    __syncthreads();

    int mt = (blockIdx.x * blockDim.x + threadIdx.x) >> 5;   // global wave id = M tile
    if (mt < nMTiles) {
        int lane = threadIdx.x & 31;
        int g  = lane >> 4;
        int ln = lane & 15;
        int row = mt * 16 + ln;

        v8f acc[NT];
#pragma unroll
        for (int t = 0; t < NT; ++t) acc[t] = (v8f){};

#pragma unroll
        for (int kbi = 0; kbi < KT; ++kbi) {
            // A fragment: two contiguous 8-float chunks -> 16 bf16
            v16bf a;
            const float* ar = A + (size_t)row * K + kbi * 32 + 8 * g;
#pragma unroll
            for (int i = 0; i < 8; ++i) a[i] = (__bf16)ar[i];
#pragma unroll
            for (int i = 0; i < 8; ++i) a[8 + i] = (__bf16)ar[16 + i];

#pragma unroll
            for (int t = 0; t < NT; ++t) {
                int f = kbi * NT + t;
                v8bf lo = *(const v8bf*)(sW + (size_t)f * 512 + lane * 8);
                v8bf hi = *(const v8bf*)(sW + (size_t)f * 512 + 256 + lane * 8);
                v16bf b;
#pragma unroll
                for (int i = 0; i < 8; ++i) b[i] = lo[i];
#pragma unroll
                for (int i = 0; i < 8; ++i) b[8 + i] = hi[i];
                acc[t] = __builtin_amdgcn_wmma_f32_16x16x32_bf16(
                             false, a, false, b, (short)0, acc[t], false, false);
            }
        }

#pragma unroll
        for (int t = 0; t < NT; ++t) {
            float* crow = C + (size_t)(mt * 16 + 8 * g) * NCOLS + t * 16 + ln;
#pragma unroll
            for (int r = 0; r < 8; ++r) crow[(size_t)r * NCOLS] = acc[t][r];
        }
    }
}

// ---------------------------------------------------------------------------
// Edge aggregation: agg[dst] += h[src] * dinv[src]*dinv[dst].
// One wave per edge; lane handles F/32 CONSECUTIVE floats (-> b128/b64
// gathers, 512B/256B coalesced per wave).  h is L2-resident (51.2 / 25.6 MB
// << 192 MB L2) so gather + scatter-atomics run at L2 speed.
// ---------------------------------------------------------------------------
template <int F>
__global__ void edge_agg_kernel(const long long* __restrict__ src, const long long* __restrict__ dst,
                                const float* __restrict__ dinv, const float* __restrict__ h,
                                float* __restrict__ agg, int E) {
    constexpr int V = F / 32;                                 // floats per lane
    int wid = (blockIdx.x * blockDim.x + threadIdx.x) >> 5;
    if (wid >= E) return;                                     // whole-wave exit
    int lane = threadIdx.x & 31;
    int s = (int)src[wid];
    int d = (int)dst[wid];
    float coef = dinv[s] * dinv[d];
    const float* hr = h + (size_t)s * F + lane * V;
    float* ar = agg + (size_t)d * F + lane * V;
    float v[V];
#pragma unroll
    for (int j = 0; j < V; ++j) v[j] = hr[j];
#pragma unroll
    for (int j = 0; j < V; ++j) atomic_add_f32(&ar[j], v[j] * coef);
}

// ---------------------------------------------------------------------------
// Layer-1 finalize: agg = relu(agg + h*dinv^2 + b)
// ---------------------------------------------------------------------------
__global__ void finalize_relu(const float* __restrict__ h, const float* __restrict__ dinv,
                              const float* __restrict__ bias, float* __restrict__ agg,
                              int n, int F) {
    int t = blockIdx.x * blockDim.x + threadIdx.x;
    if (t >= n * F) return;
    int node = t / F;
    int f    = t - node * F;
    float di = dinv[node];
    float v  = agg[t] + h[t] * di * di + bias[f];
    agg[t]   = v > 0.0f ? v : 0.0f;
}

// ---------------------------------------------------------------------------
// Layer-2 finalize + log_softmax over 64 features.  One wave per node,
// 2 features per lane, wave32 shuffle reductions.
// ---------------------------------------------------------------------------
__global__ void finalize_logsoftmax(const float* __restrict__ h2, const float* __restrict__ dinv,
                                    const float* __restrict__ bias, float* __restrict__ out, int n) {
    int node = (blockIdx.x * blockDim.x + threadIdx.x) >> 5;
    if (node >= n) return;
    int lane = threadIdx.x & 31;
    float di = dinv[node];
    float dd = di * di;
    size_t base = (size_t)node * 64;
    int f0 = lane * 2, f1 = lane * 2 + 1;
    float v0 = out[base + f0] + h2[base + f0] * dd + bias[f0];
    float v1 = out[base + f1] + h2[base + f1] * dd + bias[f1];

    float m = fmaxf(v0, v1);
#pragma unroll
    for (int o = 16; o > 0; o >>= 1) m = fmaxf(m, __shfl_xor(m, o, 32));
    float s = __expf(v0 - m) + __expf(v1 - m);
#pragma unroll
    for (int o = 16; o > 0; o >>= 1) s += __shfl_xor(s, o, 32);
    float lse = __logf(s);
    out[base + f0] = v0 - m - lse;
    out[base + f1] = v1 - m - lse;
}

// ---------------------------------------------------------------------------
// Launch
// ---------------------------------------------------------------------------
extern "C" void kernel_launch(void* const* d_in, const int* in_sizes, int n_in,
                              void* d_out, int out_size, void* d_ws, size_t ws_size,
                              hipStream_t stream) {
    (void)n_in; (void)out_size; (void)ws_size;
    const float*     x   = (const float*)d_in[0];       // [n, 256]
    const long long* ei  = (const long long*)d_in[1];   // [2, E] int64
    const float*     W1  = (const float*)d_in[2];       // [256, 128]
    const float*     b1  = (const float*)d_in[3];       // [128]
    const float*     W2  = (const float*)d_in[4];       // [128, 64]
    const float*     b2  = (const float*)d_in[5];       // [64]
    float*           out = (float*)d_out;               // [n, 64]

    const int n = in_sizes[0] / 256;
    const int E = in_sizes[1] / 2;
    const long long* src = ei;
    const long long* dst = ei + E;

    // Workspace (floats): dinv | h1 (n*128) | agg1 (n*128) | wp1 | wp2
    float* ws   = (float*)d_ws;
    float* dinv = ws;
    float* h1   = ws + (size_t)((n + 255) / 256) * 256;    // 1KB-aligned
    float* agg1 = h1 + (size_t)n * 128;
    __bf16* wp1 = (__bf16*)(agg1 + (size_t)n * 128);       // 256*128 bf16 = 64KB
    __bf16* wp2 = wp1 + 256 * 128;                         // 128*64  bf16 = 16KB
    float* h2   = h1;                                      // reuse after finalize1

    hipMemsetAsync(dinv, 0, (size_t)n * sizeof(float), stream);
    hipMemsetAsync(agg1, 0, (size_t)n * 128 * sizeof(float), stream);
    hipMemsetAsync(out,  0, (size_t)n * 64  * sizeof(float), stream);

    degree_kernel<<<(E + 255) / 256, 256, 0, stream>>>(dst, dinv, E);
    dinv_kernel<<<(n + 255) / 256, 256, 0, stream>>>(dinv, n);

    pack_w<8><<<(256 * 128 + 255) / 256, 256, 0, stream>>>(W1, wp1, 256);
    pack_w<4><<<(128 * 64 + 255) / 256, 256, 0, stream>>>(W2, wp2, 128);

    const int mTiles = (n + 15) / 16;          // 6250 (exact: n = 100000)

    // Layer 1: h1 = x @ W1  (K=256, N=128; one wave per 16-row tile)
    gemm_wmma_bf16<256, 128><<<(mTiles + 7) / 8, 256, 0, stream>>>(x, wp1, h1, mTiles);
    edge_agg_kernel<128><<<(E + 7) / 8, 256, 0, stream>>>(src, dst, dinv, h1, agg1, E);
    finalize_relu<<<((size_t)n * 128 + 255) / 256, 256, 0, stream>>>(h1, dinv, b1, agg1, n, 128);

    // Layer 2: h2 = agg1 @ W2  (K=128, N=64)
    gemm_wmma_bf16<128, 64><<<(mTiles + 7) / 8, 256, 0, stream>>>(agg1, wp2, h2, mTiles);
    edge_agg_kernel<64><<<(E + 7) / 8, 256, 0, stream>>>(src, dst, dinv, h2, out, E);
    finalize_logsoftmax<<<(n + 7) / 8, 256, 0, stream>>>(h2, dinv, b2, out, n);
}
